// PointNet_2010044694955
// MI455X (gfx1250) — matrix operand order repaired and checked by reading, not compiled, not run on hardware
//
#include <hip/hip_runtime.h>

// DGCNN PointNet for MI455X (gfx1250, wave32).
// - knn Gram matrices (dominant compute) : V_WMMA_F32_16X16X4_F32 (exact f32),
//   column tiles staged to LDS by the Tensor Data Mover (double-buffered DMA).
// - EdgeConv linear layers               : V_WMMA_F32_16X16X32_F16 (f16 in, f32 acc)
// - tiny head (pool/BN/MLP)              : scalar f32 (sub-1% of FLOPs)

#define BB 64
#define NN 1024
#define FF 5
#define WW 128
#define DD2 768

typedef __attribute__((ext_vector_type(16))) _Float16 v16h;
typedef __attribute__((ext_vector_type(8)))  float    v8f;
typedef __attribute__((ext_vector_type(2)))  float    v2f;
typedef __attribute__((ext_vector_type(4)))  unsigned int u32x4;
typedef __attribute__((ext_vector_type(8)))  int      i32x8;
typedef __attribute__((ext_vector_type(4)))  int      i32x4;

__device__ __forceinline__ v8f wmma_f16(v16h a, v16h b, v8f c) {
  return __builtin_amdgcn_wmma_f32_16x16x32_f16(false, a, false, b, (short)0, c, false, false);
}
__device__ __forceinline__ v8f wmma_f32(v2f a, v2f b, v8f c) {
  return __builtin_amdgcn_wmma_f32_16x16x4_f32(false, a, false, b, (short)0, c, false, false);
}

// ---- fragment loaders (A x B^T, both operands row-major) -------------------
// A 16x32 f16: lane L = row (L&15); halves {k0+(L>>4)*8 .. +7} and {+16 .. +23}
__device__ __forceinline__ v16h load_a16(const _Float16* base, int ld, int row0, int k0, int lane) {
  const _Float16* p = base + (size_t)(row0 + (lane & 15)) * ld + k0 + ((lane >> 4) << 3);
  union { v16h v; float4 q[2]; } u;
  u.q[0] = *(const float4*)(p);
  u.q[1] = *(const float4*)(p + 16);
  return u.v;
}
// B 32x16 f16: lane L = col (L&15); halves {k0+(L>>4)*8 .. +7} and {+16 .. +23}
__device__ __forceinline__ v16h load_b16(const _Float16* base, int ld, int col0, int k0, int lane) {
  const _Float16* p = base + (size_t)(col0 + (lane & 15)) * ld + k0 + ((lane >> 4) << 3);
  union { v16h v; float4 q[2]; } u;
  u.q[0] = *(const float4*)(p);
  u.q[1] = *(const float4*)(p + 16);
  return u.v;
}
// A 16x4 / B 4x16 f32: lane L = row/col (L&15); float pair at k0+(L>>4)*2
__device__ __forceinline__ v2f load_ab4(const float* base, int ld, int rc0, int k0, int lane) {
  const float* p = base + (size_t)(rc0 + (lane & 15)) * ld + k0 + ((lane >> 4) << 1);
  return *(const v2f*)p;
}

// ---- TDM: DMA a 16-row x 128-col f32 tile (row stride 128) into LDS --------
// D# group0: count=1(valid user desc), lds_addr, global_addr[56:0], type=2.
// D# group1: data_size=4B; tensor_dim0=128, tensor_dim1=16; tile 128x16;
//            tensor_dim0_stride=128 elements.
__device__ __forceinline__ void tdm_load_tile(const float* gsrc, unsigned lds_off) {
  unsigned long long ga = (unsigned long long)(uintptr_t)gsrc;
  u32x4 g0 = { 1u, lds_off, (unsigned)(ga & 0xffffffffu),
               (unsigned)((ga >> 32) & 0x01ffffffu) | (2u << 30) };
  i32x8 g1 = { 0x20000,          // data_size = 2 (4 bytes)
               (int)(128u << 16),// tensor_dim0[15:0] = 128 (bits 79:48)
               (int)(16u << 16), // tensor_dim1[15:0] = 16  (bits 111:80)
               (int)(128u << 16),// tile_dim0 = 128 (bits 127:112)
               16,               // tile_dim1 = 16  (bits 143:128)
               128,              // tensor_dim0_stride[31:0] = 128 (bits 207:160)
               0, 0 };
  i32x4 z4 = { 0, 0, 0, 0 };
#if defined(__clang_major__) && __clang_major__ >= 23
  i32x8 z8 = { 0, 0, 0, 0, 0, 0, 0, 0 };
  __builtin_amdgcn_tensor_load_to_lds(g0, g1, z4, z4, z8, 0);
#else
  __builtin_amdgcn_tensor_load_to_lds(g0, g1, z4, z4, 0);
#endif
}

// ---- weight prep: transpose to out-major f16 -------------------------------
__global__ void prep_kernel(const float* __restrict__ p1w2, const float* __restrict__ cw,
                            _Float16* __restrict__ w2t, _Float16* __restrict__ cwt) {
  int t = blockIdx.x * blockDim.x + threadIdx.x;
  if (t < 128 * 128) {                       // w2t[o][i] = p1_w2[i][o]
    int o = t >> 7, i = t & 127;
    w2t[t] = (_Float16)p1w2[i * 128 + o];
  }
  if (t < 2 * 128 * 256) {                   // cwt[f][o][i] = c_w[f][i][o]
    int f = t >> 15;
    int r = t & 32767;
    int o = r >> 8, i = r & 255;
    cwt[t] = (_Float16)cw[f * 256 * 128 + i * 128 + o];
  }
}

// ---- knn on raw 5-dim input (scalar; self is always neighbor 0) ------------
__global__ void knn_x_kernel(const float* __restrict__ x, int* __restrict__ nbr) {
  __shared__ float xs[NN * FF];
  int b = blockIdx.x >> 2;
  int n = ((blockIdx.x & 3) << 8) + threadIdx.x;
  const float* xb = x + (size_t)b * NN * FF;
  for (int i = threadIdx.x; i < NN * FF; i += 256) xs[i] = xb[i];
  __syncthreads();
  float xi[FF];
  for (int f = 0; f < FF; ++f) xi[f] = xs[n * FF + f];
  float best = 3.4e38f; int bi = n;
  for (int j = 0; j < NN; ++j) {
    float d = 0.f;
    for (int f = 0; f < FF; ++f) { float t = xs[j * FF + f] - xi[f]; d += t * t; }
    if (j != n && d < best) { best = d; bi = j; }
  }
  nbr[(size_t)b * NN + n] = bi;
}

// ---- EdgeConv1: MLP(10->128->128), ReLU, max over {self, nbr} --------------
__global__ void edgeconv1_kernel(const float* __restrict__ x, const int* __restrict__ nbr,
                                 const float* __restrict__ w1, const float* __restrict__ b1,
                                 const _Float16* __restrict__ w2t, const float* __restrict__ b2,
                                 float* __restrict__ y) {
  __shared__ __align__(16) float EF[32 * 10];
  __shared__ __align__(16) _Float16 H[32 * 128];  // rows 0-15: self edge, 16-31: nbr edge
  int b = blockIdx.x >> 6;
  int p0 = (blockIdx.x & 63) << 4;
  int tid = threadIdx.x;
  for (int i = tid; i < 32 * 10; i += 128) {
    int r = i / 10, f = i % 10;
    int p = p0 + (r & 15);
    const float* xp = x + ((size_t)b * NN + p) * FF;
    float v;
    if (f < FF) v = xp[f];
    else if (r < 16) v = 0.f;
    else {
      int j = nbr[(size_t)b * NN + p];
      v = x[((size_t)b * NN + j) * FF + (f - FF)] - xp[f - FF];
    }
    EF[i] = v;
  }
  __syncthreads();
  for (int i = tid; i < 32 * 128; i += 128) {   // layer 1 (tiny K=10): scalar
    int r = i >> 7, w = i & 127;
    float acc = b1[w];
    for (int f = 0; f < 10; ++f) acc += EF[r * 10 + f] * w1[f * 128 + w];
    H[i] = (_Float16)(acc > 0.f ? acc : 0.f);
  }
  __syncthreads();
  int wave = tid >> 5, lane = tid & 31;
  for (int t = 0; t < 2; ++t) {                 // layer 2: WMMA f16 (K=128)
    int ct = wave * 2 + t;
    v8f acc0 = {}, acc1 = {};
#pragma unroll
    for (int kc = 0; kc < 4; ++kc) {
      v16h bf = load_b16(w2t, 128, ct * 16, kc * 32, lane);
      v16h a0 = load_a16(H, 128, 0, kc * 32, lane);
      v16h a1 = load_a16(H, 128, 16, kc * 32, lane);
      acc0 = wmma_f16(a0, bf, acc0);
      acc1 = wmma_f16(a1, bf, acc1);
    }
    int o = ct * 16 + (lane & 15);
    float bias = b2[o];
    int mbase = p0 + ((lane >> 4) << 3);
#pragma unroll
    for (int e = 0; e < 8; ++e) {
      int p = mbase + e;
      float s = acc0[e] + bias; s = s > 0.f ? s : 0.f;
      float q = acc1[e] + bias; q = q > 0.f ? q : 0.f;
      y[((size_t)b * NN + p) * WW + o] = s > q ? s : q;
    }
  }
}

// ---- squared norms ----------------------------------------------------------
__global__ void sqnorm_kernel(const float* __restrict__ y, float* __restrict__ sq) {
  int t = blockIdx.x * blockDim.x + threadIdx.x;
  const float* p = y + (size_t)t * WW;
  float s = 0.f;
  for (int c = 0; c < WW; ++c) { float v = p[c]; s += v * v; }
  sq[t] = s;
}

// ---- knn on 128-dim features ------------------------------------------------
// Gram matrix via f32 WMMA; 16x128 column tiles DMA'd to LDS by the TDM,
// double-buffered so the copy of tile ct+1 overlaps the WMMAs on tile ct.
// 4 waves/WG share one batch and the staged tiles; each wave owns 16 rows.
__global__ void knn128_kernel(const float* __restrict__ y, const float* __restrict__ sq,
                              int* __restrict__ nbr) {
  __shared__ __align__(16) float Bt[2][16 * WW];   // 2 x 8KB tile buffers
  int wave = threadIdx.x >> 5;
  int lane = threadIdx.x & 31;
  int b = blockIdx.x >> 4;                         // 16 WGs per batch
  int row0 = (((blockIdx.x & 15) << 2) + wave) << 4;
  const float* Y = y + (size_t)b * NN * WW;
  const float* SQ = sq + (size_t)b * NN;
  v2f a[32];
#pragma unroll
  for (int kc = 0; kc < 32; ++kc) a[kc] = load_ab4(Y, WW, row0, kc * 4, lane);
  int mb = (lane >> 4) << 3;
  float sqr[8], bestd[8]; int bestj[8];
#pragma unroll
  for (int e = 0; e < 8; ++e) { sqr[e] = SQ[row0 + mb + e]; bestd[e] = 3.4e38f; bestj[e] = NN; }

  unsigned lds0 = (unsigned)(uintptr_t)&Bt[0][0];
  unsigned lds1 = (unsigned)(uintptr_t)&Bt[1][0];
  if (wave == 0) tdm_load_tile(Y, lds0);           // prime tile 0
  for (int ct = 0; ct < 64; ++ct) {
    int buf = ct & 1;
    if (wave == 0) {
      if (ct + 1 < 64) {                           // prefetch next tile, then wait
        tdm_load_tile(Y + (size_t)(ct + 1) * 16 * WW, buf ? lds0 : lds1);
        __builtin_amdgcn_s_wait_tensorcnt(1);      // in-order: tile ct resident
      } else {
        __builtin_amdgcn_s_wait_tensorcnt(0);
      }
    }
    __syncthreads();                               // RAW: DMA writes -> all waves read
    const float* Bb = &Bt[buf][0];
    v8f acc = {};
#pragma unroll
    for (int kc = 0; kc < 32; ++kc) {
      v2f bf = load_ab4(Bb, WW, 0, kc * 4, lane);
      acc = wmma_f32(a[kc], bf, acc);
    }
    int n = ct * 16 + (lane & 15);
    float sqc = SQ[n];
#pragma unroll
    for (int e = 0; e < 8; ++e) {
      int m = row0 + mb + e;
      float d = sqr[e] + sqc - 2.f * acc[e];
      if (m != n && (d < bestd[e] || (d == bestd[e] && n < bestj[e]))) { bestd[e] = d; bestj[e] = n; }
    }
    __syncthreads();                               // WAR: reads done before next DMA reuses buf
  }
  for (int msk = 1; msk <= 8; msk <<= 1) {         // fold across the 16 lanes per row-half
#pragma unroll
    for (int e = 0; e < 8; ++e) {
      float od = __shfl_xor(bestd[e], msk);
      int   oj = __shfl_xor(bestj[e], msk);
      if (od < bestd[e] || (od == bestd[e] && oj < bestj[e])) { bestd[e] = od; bestj[e] = oj; }
    }
  }
  if ((lane & 15) == 0)
#pragma unroll
    for (int e = 0; e < 8; ++e) nbr[(size_t)b * NN + row0 + mb + e] = bestj[e];
}

// ---- EdgeConv2/3: Linear(256->128)+ReLU, max over {self, nbr} --------------
__global__ void edgeconv23_kernel(const float* __restrict__ yin, const int* __restrict__ nbr,
                                  const _Float16* __restrict__ wt, const float* __restrict__ bias,
                                  float* __restrict__ yout) {
  __shared__ __align__(16) _Float16 EF[32 * 256];  // 16 KB edge-feature stage
  int b = blockIdx.x >> 6;
  int p0 = (blockIdx.x & 63) << 4;
  int tid = threadIdx.x;
  for (int i = tid; i < 32 * 256; i += 128) {
    int r = i >> 8, c = i & 255;
    int p = p0 + (r & 15);
    float v;
    if (c < WW) v = yin[((size_t)b * NN + p) * WW + c];
    else if (r < 16) v = 0.f;
    else {
      int j = nbr[(size_t)b * NN + p];
      v = yin[((size_t)b * NN + j) * WW + (c - WW)] - yin[((size_t)b * NN + p) * WW + (c - WW)];
    }
    EF[i] = (_Float16)v;
  }
  __syncthreads();
  int wave = tid >> 5, lane = tid & 31;
  for (int t = 0; t < 2; ++t) {
    int ct = wave * 2 + t;
    v8f acc0 = {}, acc1 = {};
#pragma unroll
    for (int kc = 0; kc < 8; ++kc) {
      v16h bf = load_b16(wt, 256, ct * 16, kc * 32, lane);
      v16h a0 = load_a16(EF, 256, 0, kc * 32, lane);
      v16h a1 = load_a16(EF, 256, 16, kc * 32, lane);
      acc0 = wmma_f16(a0, bf, acc0);
      acc1 = wmma_f16(a1, bf, acc1);
    }
    int o = ct * 16 + (lane & 15);
    float bv = bias[o];
    int mbase = p0 + ((lane >> 4) << 3);
#pragma unroll
    for (int e = 0; e < 8; ++e) {
      int p = mbase + e;
      float s = acc0[e] + bv; s = s > 0.f ? s : 0.f;
      float q = acc1[e] + bv; q = q > 0.f ? q : 0.f;
      yout[((size_t)b * NN + p) * WW + o] = s > q ? s : q;
    }
  }
}

// ---- global mean/max pool ---------------------------------------------------
__global__ void pool_kernel(const float* __restrict__ y1, const float* __restrict__ y2,
                            const float* __restrict__ y3, float* __restrict__ g) {
  int t = blockIdx.x * blockDim.x + threadIdx.x;   // B * 384
  int b = t / 384, f = t % 384;
  const float* base = (f < 128) ? y1 : (f < 256 ? y2 : y3);
  const float* src = base + ((size_t)b * NN) * WW + (f & 127);
  float s = 0.f, mx = -3.4e38f;
  for (int n = 0; n < NN; ++n) {
    float v = src[(size_t)n * WW];
    s += v; mx = mx > v ? mx : v;
  }
  g[b * DD2 + f] = s * (1.f / NN);
  g[b * DD2 + 384 + f] = mx;
}

// ---- BatchNorm (batch statistics, biased var) ------------------------------
__global__ void bn_kernel(const float* __restrict__ g, const float* __restrict__ gam,
                          const float* __restrict__ bet, float* __restrict__ h) {
  int f = blockIdx.x * blockDim.x + threadIdx.x;
  if (f >= DD2) return;
  float s = 0.f;
  for (int b = 0; b < BB; ++b) s += g[b * DD2 + f];
  float mu = s * (1.f / BB);
  float v = 0.f;
  for (int b = 0; b < BB; ++b) { float d = g[b * DD2 + f] - mu; v += d * d; }
  float rs = rsqrtf(v * (1.f / BB) + 1e-5f);
  float ga = gam[f], be = bet[f];
  for (int b = 0; b < BB; ++b) h[b * DD2 + f] = (g[b * DD2 + f] - mu) * rs * ga + be;
}

// ---- Linear 768->768 + LeakyReLU -------------------------------------------
__global__ void linear_kernel(const float* __restrict__ hin, const float* __restrict__ W,
                              const float* __restrict__ bias, float* __restrict__ hout) {
  int t = blockIdx.x * blockDim.x + threadIdx.x;   // B * 768
  int b = t / DD2, o = t % DD2;
  const float* hb = hin + b * DD2;
  float acc = bias[o];
  for (int i = 0; i < DD2; ++i) acc += hb[i] * W[i * DD2 + o];
  hout[t] = acc > 0.f ? acc : acc * 0.01f;
}

// ---- output head ------------------------------------------------------------
__global__ void out_kernel(const float* __restrict__ h, const float* __restrict__ ow,
                           const float* __restrict__ ob, float* __restrict__ out) {
  int b = threadIdx.x;
  if (b >= BB) return;
  const float* hb = h + b * DD2;
  float acc = ob[0];
  for (int i = 0; i < DD2; ++i) acc += hb[i] * ow[i];
  out[b] = acc;
}

extern "C" void kernel_launch(void* const* d_in, const int* in_sizes, int n_in,
                              void* d_out, int out_size, void* d_ws, size_t ws_size,
                              hipStream_t stream) {
  const float* x    = (const float*)d_in[0];
  const float* p1w1 = (const float*)d_in[1];
  const float* p1b1 = (const float*)d_in[2];
  const float* p1w2 = (const float*)d_in[3];
  const float* p1b2 = (const float*)d_in[4];
  const float* cw   = (const float*)d_in[5];
  const float* cb   = (const float*)d_in[6];
  const float* bng  = (const float*)d_in[7];
  const float* bnb  = (const float*)d_in[8];
  const float* linw = (const float*)d_in[9];
  const float* linb = (const float*)d_in[10];
  const float* outw = (const float*)d_in[11];
  const float* outb = (const float*)d_in[12];
  float* out = (float*)d_out;

  char* ws = (char*)d_ws;
  auto alloc = [&](size_t bytes) {
    char* p = ws;
    ws += (bytes + 255) & ~(size_t)255;
    return p;
  };
  int*      nbr = (int*)     alloc((size_t)BB * NN * 4);
  float*    y1  = (float*)   alloc((size_t)BB * NN * WW * 4);
  float*    y2  = (float*)   alloc((size_t)BB * NN * WW * 4);
  float*    y3  = (float*)   alloc((size_t)BB * NN * WW * 4);
  float*    sq  = (float*)   alloc((size_t)BB * NN * 4);
  float*    g   = (float*)   alloc((size_t)BB * DD2 * 4);
  float*    hA  = (float*)   alloc((size_t)BB * DD2 * 4);
  float*    hB  = (float*)   alloc((size_t)BB * DD2 * 4);
  _Float16* w2t = (_Float16*)alloc((size_t)128 * 128 * 2);
  _Float16* cwt = (_Float16*)alloc((size_t)2 * 128 * 256 * 2);

  prep_kernel<<<256, 256, 0, stream>>>(p1w2, cw, w2t, cwt);
  knn_x_kernel<<<BB * 4, 256, 0, stream>>>(x, nbr);
  edgeconv1_kernel<<<BB * 64, 128, 0, stream>>>(x, nbr, p1w1, p1b1, w2t, p1b2, y1);

  sqnorm_kernel<<<BB * NN / 256, 256, 0, stream>>>(y1, sq);
  knn128_kernel<<<BB * 16, 128, 0, stream>>>(y1, sq, nbr);
  edgeconv23_kernel<<<BB * 64, 128, 0, stream>>>(y1, nbr, cwt, cb, y2);

  sqnorm_kernel<<<BB * NN / 256, 256, 0, stream>>>(y2, sq);
  knn128_kernel<<<BB * 16, 128, 0, stream>>>(y2, sq, nbr);
  edgeconv23_kernel<<<BB * 64, 128, 0, stream>>>(y2, nbr, cwt + 128 * 256, cb + 128, y3);

  pool_kernel<<<BB * 384 / 256, 256, 0, stream>>>(y1, y2, y3, g);
  bn_kernel<<<3, 256, 0, stream>>>(g, bng, bnb, hA);

  float* src = hA;
  float* dst = hB;
  for (int l = 0; l < 5; ++l) {
    linear_kernel<<<BB * DD2 / 256, 256, 0, stream>>>(src, linw + (size_t)l * DD2 * DD2,
                                                      linb + (size_t)l * DD2, dst);
    float* t = src; src = dst; dst = t;
  }
  out_kernel<<<1, 64, 0, stream>>>(src, outw, outb, out);
}
